// RelativePositionMultiheadSelfAttention_15882789061110
// MI455X (gfx1250) — matrix operand order repaired and checked by reading, not compiled, not run on hardware
//
#include <hip/hip_runtime.h>

// ---------------------------------------------------------------------------
// Relative-position multi-head self-attention for MI455X (gfx1250, wave32).
// All GEMMs use v_wmma_f32_16x16x32_bf16 (fp32 accumulate).
// ---------------------------------------------------------------------------

#define DMODEL 1024
#define NHEADS 16
#define HD     64
#define LSEQ   1024
#define BATCH  4
#define MAXD   128
#define RCOLS  272   // 257 rel-pos entries padded to 17*16

typedef __attribute__((ext_vector_type(16))) __bf16 v16bf;
typedef __attribute__((ext_vector_type(8)))  __bf16 bf16x8;
typedef __attribute__((ext_vector_type(8)))  float  v8f;

static __device__ __forceinline__ unsigned short f2bf(float f) {
  unsigned int u = __float_as_uint(f);
  u += 0x7fffu + ((u >> 16) & 1u);          // round-to-nearest-even
  return (unsigned short)(u >> 16);
}

static __device__ __forceinline__ v16bf make16(bf16x8 lo, bf16x8 hi) {
  union { bf16x8 h[2]; v16bf v; } u;
  u.h[0] = lo; u.h[1] = hi;
  return u.v;
}

// A fragment (16x32 bf16) from row-major [M x K] src, row stride ld.
// ISA layout: lane half=l>>4, m=l&15; elem j -> K = (j/8)*16 + half*8 + j%8.
static __device__ __forceinline__ v16bf load_a(const __bf16* src, int ld, int k0, int lane) {
  int m = lane & 15, half = lane >> 4;
  const __bf16* p = src + (size_t)m * ld + k0 + half * 8;
  bf16x8 lo = *(const bf16x8*)(p);
  bf16x8 hi = *(const bf16x8*)(p + 16);
  return make16(lo, hi);
}

// B fragment (32x16 bf16): B[kk][n] = srcT[n][kk]; srcT row-major [N x K], stride ld.
// ISA layout: lane n=l&15, half=l>>4; elem j -> K = (j/8)*16 + half*8 + j%8.
static __device__ __forceinline__ v16bf load_b(const __bf16* srcT, int ld, int n0, int k0, int lane) {
  int n = n0 + (lane & 15), half = lane >> 4;
  const __bf16* p = srcT + (size_t)n * ld + k0 + half * 8;
  bf16x8 lo = *(const bf16x8*)(p);
  bf16x8 hi = *(const bf16x8*)(p + 16);
  return make16(lo, hi);
}

static __device__ __forceinline__ v8f wmma_bf16(v16bf a, v16bf b, v8f c) {
  return __builtin_amdgcn_wmma_f32_16x16x32_bf16(false, a, false, b, (short)0, c, false, false);
}

static __device__ __forceinline__ v8f zero8() {
  v8f c;
#pragma unroll
  for (int i = 0; i < 8; ++i) c[i] = 0.0f;
  return c;
}

// ---------------------------------------------------------------------------
// Precision / layout conversion kernels
// ---------------------------------------------------------------------------
__global__ void cvt_bf16_kernel(const float* __restrict__ in,
                                unsigned short* __restrict__ out, int nElem) {
  int i = blockIdx.x * blockDim.x + threadIdx.x;
  if (i < nElem) out[i] = f2bf(in[i]);
}

// W[k][n] row-major -> WT[n][k] bf16 (so B-fragment reads are contiguous)
__global__ void transpose_bf16_kernel(const float* __restrict__ W,
                                      unsigned short* __restrict__ WT) {
  int i = blockIdx.x * blockDim.x + threadIdx.x;   // 1M elements
  int k = i >> 10, n = i & 1023;                   // coalesced read over n
  WT[(size_t)n * DMODEL + k] = f2bf(W[i]);
}

__global__ void cvt_pos_kernel(const float* __restrict__ pos,
                               unsigned short* __restrict__ posBf) {
  int i = blockIdx.x * blockDim.x + threadIdx.x;   // RCOLS*HD
  if (i < RCOLS * HD) {
    int r = i >> 6;
    posBf[i] = (r < 2 * MAXD + 1) ? f2bf(pos[i]) : (unsigned short)0;
  }
}

// ---------------------------------------------------------------------------
// QKV projection: [4096 x 1024] @ WT -> q/k head-split [BH][L][HD], v transposed
// [BH][HD][L]. 4 waves/block, each wave: 16x64 output tile, K-loop of 32.
// ---------------------------------------------------------------------------
__global__ __launch_bounds__(128)
void gemm_qkv_kernel(const unsigned short* __restrict__ qry_,
                     const unsigned short* __restrict__ wqT_,
                     const unsigned short* __restrict__ wkT_,
                     const unsigned short* __restrict__ wvT_,
                     unsigned short* __restrict__ qh,
                     unsigned short* __restrict__ kh,
                     unsigned short* __restrict__ vhT) {
  int lane = threadIdx.x & 31, wave = threadIdx.x >> 5;
  int m0 = blockIdx.x * 64 + wave * 16;
  int n0 = blockIdx.y * 64;
  int which = blockIdx.z;
  const __bf16* A  = (const __bf16*)qry_ + (size_t)m0 * DMODEL;
  const __bf16* WT = (const __bf16*)(which == 0 ? wqT_ : which == 1 ? wkT_ : wvT_);

  v8f acc[4];
#pragma unroll
  for (int nt = 0; nt < 4; ++nt) acc[nt] = zero8();

  for (int k0 = 0; k0 < DMODEL; k0 += 32) {
    v16bf a = load_a(A, DMODEL, k0, lane);
#pragma unroll
    for (int nt = 0; nt < 4; ++nt)
      acc[nt] = wmma_bf16(a, load_b(WT, DMODEL, n0 + nt * 16, k0, lane), acc[nt]);
  }

  int half = lane >> 4, n = lane & 15;
#pragma unroll
  for (int nt = 0; nt < 4; ++nt) {
#pragma unroll
    for (int r = 0; r < 8; ++r) {
      int gr = m0 + r + 8 * half;        // token row in [0,4096)
      int gc = n0 + nt * 16 + n;         // model dim in [0,1024)
      int bb = gr >> 10, lpos = gr & 1023;
      int hh = gc >> 6,  hd   = gc & 63;
      unsigned short v = f2bf(acc[nt][r]);
      size_t bh = (size_t)(bb * NHEADS + hh);
      if (which == 0)      qh [(bh * LSEQ + lpos) * HD + hd] = v;
      else if (which == 1) kh [(bh * LSEQ + lpos) * HD + hd] = v;
      else                 vhT[(bh * HD   + hd)  * LSEQ + lpos] = v;
    }
  }
}

// ---------------------------------------------------------------------------
// Fused attention: one wave per (b,h,q-tile of 16).
//  1. relrow[16][272] = q_tile @ pos_emb^T via WMMA into LDS
//  2. flash-style loop over 32 keys/step: QK^T WMMA, gather rel bias from LDS,
//     online softmax (16-lane xor reductions), P tile bounced via 1KB LDS to
//     A-fragment layout, PV WMMA accumulate.
// ---------------------------------------------------------------------------
__global__ __launch_bounds__(32)
void attn_kernel(const unsigned short* __restrict__ qh_,
                 const unsigned short* __restrict__ kh_,
                 const unsigned short* __restrict__ vhT_,
                 const unsigned short* __restrict__ posBf_,
                 const float* __restrict__ bias,
                 unsigned short* __restrict__ attnBf) {
  __shared__ float relrow[16 * RCOLS];
  __shared__ __align__(16) unsigned short pbuf[16 * 32];

  int lane = threadIdx.x;
  int qtile = blockIdx.x;
  int bh = blockIdx.y;
  int b = bh >> 4, h = bh & 15;
  int q0 = qtile * 16;
  int half = lane >> 4, n = lane & 15;

  const __bf16* qB   = (const __bf16*)qh_  + (size_t)bh * LSEQ * HD;
  const __bf16* kB   = (const __bf16*)kh_  + (size_t)bh * LSEQ * HD;
  const __bf16* vT   = (const __bf16*)vhT_ + (size_t)bh * HD * LSEQ;
  const __bf16* posB = (const __bf16*)posBf_;

  // Q tile A-fragments (K = hd = 64 -> two 16x32 fragments), reused everywhere.
  v16bf a0 = load_a(qB + (size_t)q0 * HD, HD, 0, lane);
  v16bf a1 = load_a(qB + (size_t)q0 * HD, HD, 32, lane);

  // relrow = q_tile @ pos_emb^T  (16 x 272)
  for (int nt = 0; nt < 17; ++nt) {
    v8f c = zero8();
    c = wmma_bf16(a0, load_b(posB, HD, nt * 16, 0, lane), c);
    c = wmma_bf16(a1, load_b(posB, HD, nt * 16, 32, lane), c);
#pragma unroll
    for (int r = 0; r < 8; ++r)
      relrow[(r + 8 * half) * RCOLS + nt * 16 + n] = c[r];
  }
  __syncthreads();

  float rmax[8], rsum[8];
#pragma unroll
  for (int r = 0; r < 8; ++r) { rmax[r] = -3.0e38f; rsum[r] = 0.0f; }
  v8f o[4];
#pragma unroll
  for (int nt = 0; nt < 4; ++nt) o[nt] = zero8();

  const float* biasRow = bias + (size_t)b * LSEQ;

  for (int k0 = 0; k0 < LSEQ; k0 += 32) {
    // content scores: two 16x16 S tiles (cols k0..k0+15, k0+16..k0+31)
    v8f s[2];
#pragma unroll
    for (int t = 0; t < 2; ++t) {
      v8f c = zero8();
      c = wmma_bf16(a0, load_b(kB, HD, k0 + t * 16, 0, lane), c);
      c = wmma_bf16(a1, load_b(kB, HD, k0 + t * 16, 32, lane), c);
      s[t] = c;
    }
    float bv0 = biasRow[k0 + n];
    float bv1 = biasRow[k0 + 16 + n];

    float p0[8], p1[8], tm[8];
#pragma unroll
    for (int r = 0; r < 8; ++r) {
      int M = r + 8 * half;
      int qg = q0 + M;
      int d0 = (k0 + n) - qg;       d0 = d0 < -MAXD ? -MAXD : (d0 > MAXD ? MAXD : d0);
      int d1 = (k0 + 16 + n) - qg;  d1 = d1 < -MAXD ? -MAXD : (d1 > MAXD ? MAXD : d1);
      float r0 = relrow[M * RCOLS + d0 + MAXD];
      float r1 = relrow[M * RCOLS + d1 + MAXD];
      p0[r] = (s[0][r] + r0) * 0.125f + bv0;   // 1/sqrt(64) = 0.125
      p1[r] = (s[1][r] + r1) * 0.125f + bv1;
      tm[r] = fmaxf(p0[r], p1[r]);
    }
    // row-max across the 16 lanes of each half (rows stay within a half)
#pragma unroll
    for (int r = 0; r < 8; ++r)
#pragma unroll
      for (int off = 1; off < 16; off <<= 1)
        tm[r] = fmaxf(tm[r], __shfl_xor(tm[r], off, 32));

    float alpha[8], ts[8];
#pragma unroll
    for (int r = 0; r < 8; ++r) {
      float mn = fmaxf(rmax[r], tm[r]);
      alpha[r] = __expf(rmax[r] - mn);
      rmax[r] = mn;
      p0[r] = __expf(p0[r] - mn);
      p1[r] = __expf(p1[r] - mn);
      ts[r] = p0[r] + p1[r];
    }
#pragma unroll
    for (int r = 0; r < 8; ++r)
#pragma unroll
      for (int off = 1; off < 16; off <<= 1)
        ts[r] += __shfl_xor(ts[r], off, 32);

#pragma unroll
    for (int r = 0; r < 8; ++r) {
      rsum[r] = rsum[r] * alpha[r] + ts[r];
      int M = r + 8 * half;
      pbuf[M * 32 + n]      = f2bf(p0[r]);     // P tile, row-major 16x32 bf16
      pbuf[M * 32 + 16 + n] = f2bf(p1[r]);
    }
#pragma unroll
    for (int nt = 0; nt < 4; ++nt)
#pragma unroll
      for (int r = 0; r < 8; ++r)
        o[nt][r] *= alpha[r];

    __syncthreads();
    v16bf ap = load_a((const __bf16*)pbuf, 32, 0, lane);
#pragma unroll
    for (int nt = 0; nt < 4; ++nt)
      o[nt] = wmma_bf16(ap, load_b(vT, LSEQ, nt * 16, k0, lane), o[nt]);
    __syncthreads();
  }

  // normalize and write merged-head bf16 output [4096 x 1024]
#pragma unroll
  for (int r = 0; r < 8; ++r) {
    int M = r + 8 * half;
    float inv = 1.0f / rsum[r];
    size_t row = (size_t)(b * LSEQ + q0 + M) * DMODEL + h * HD;
#pragma unroll
    for (int nt = 0; nt < 4; ++nt)
      attnBf[row + nt * 16 + n] = f2bf(o[nt][r] * inv);
  }
}

// ---------------------------------------------------------------------------
// Output projection: attn[4096 x 1024] bf16 @ WoutT -> fp32 d_out
// ---------------------------------------------------------------------------
__global__ __launch_bounds__(128)
void gemm_out_kernel(const unsigned short* __restrict__ attnBf_,
                     const unsigned short* __restrict__ woT_,
                     float* __restrict__ out) {
  int lane = threadIdx.x & 31, wave = threadIdx.x >> 5;
  int m0 = blockIdx.x * 64 + wave * 16;
  int n0 = blockIdx.y * 64;
  const __bf16* A  = (const __bf16*)attnBf_ + (size_t)m0 * DMODEL;
  const __bf16* WT = (const __bf16*)woT_;

  v8f acc[4];
#pragma unroll
  for (int nt = 0; nt < 4; ++nt) acc[nt] = zero8();

  for (int k0 = 0; k0 < DMODEL; k0 += 32) {
    v16bf a = load_a(A, DMODEL, k0, lane);
#pragma unroll
    for (int nt = 0; nt < 4; ++nt)
      acc[nt] = wmma_bf16(a, load_b(WT, DMODEL, n0 + nt * 16, k0, lane), acc[nt]);
  }

  int half = lane >> 4, n = lane & 15;
#pragma unroll
  for (int nt = 0; nt < 4; ++nt)
#pragma unroll
    for (int r = 0; r < 8; ++r)
      out[(size_t)(m0 + r + 8 * half) * DMODEL + n0 + nt * 16 + n] = acc[nt][r];
}

// ---------------------------------------------------------------------------
// Host launcher
// ---------------------------------------------------------------------------
extern "C" void kernel_launch(void* const* d_in, const int* in_sizes, int n_in,
                              void* d_out, int out_size, void* d_ws, size_t ws_size,
                              hipStream_t stream) {
  const float* query = (const float*)d_in[0];
  const float* bias  = (const float*)d_in[1];
  const float* Wq    = (const float*)d_in[2];
  const float* Wk    = (const float*)d_in[3];
  const float* Wv    = (const float*)d_in[4];
  const float* Wout  = (const float*)d_in[5];
  const float* pos   = (const float*)d_in[6];
  float* out = (float*)d_out;

  char* ws = (char*)d_ws;
  size_t off = 0;
  auto alloc = [&](size_t bytes) -> void* {
    void* p = ws + off;
    off = (off + bytes + 255) & ~(size_t)255;
    return p;
  };

  const size_t TOK = (size_t)BATCH * LSEQ;                 // 4096
  unsigned short* qryBf  = (unsigned short*)alloc(TOK * DMODEL * 2);
  unsigned short* wqT    = (unsigned short*)alloc((size_t)DMODEL * DMODEL * 2);
  unsigned short* wkT    = (unsigned short*)alloc((size_t)DMODEL * DMODEL * 2);
  unsigned short* wvT    = (unsigned short*)alloc((size_t)DMODEL * DMODEL * 2);
  unsigned short* woT    = (unsigned short*)alloc((size_t)DMODEL * DMODEL * 2);
  unsigned short* posBf  = (unsigned short*)alloc((size_t)RCOLS * HD * 2);
  unsigned short* qh     = (unsigned short*)alloc(TOK * DMODEL * 2);
  unsigned short* kh     = (unsigned short*)alloc(TOK * DMODEL * 2);
  unsigned short* vhT    = (unsigned short*)alloc(TOK * DMODEL * 2);
  unsigned short* attnBf = (unsigned short*)alloc(TOK * DMODEL * 2);
  (void)ws_size; (void)in_sizes; (void)n_in; (void)out_size;

  int nq = (int)(TOK * DMODEL);
  cvt_bf16_kernel<<<(nq + 255) / 256, 256, 0, stream>>>(query, qryBf, nq);
  transpose_bf16_kernel<<<(DMODEL * DMODEL) / 256, 256, 0, stream>>>(Wq, wqT);
  transpose_bf16_kernel<<<(DMODEL * DMODEL) / 256, 256, 0, stream>>>(Wk, wkT);
  transpose_bf16_kernel<<<(DMODEL * DMODEL) / 256, 256, 0, stream>>>(Wv, wvT);
  transpose_bf16_kernel<<<(DMODEL * DMODEL) / 256, 256, 0, stream>>>(Wout, woT);
  cvt_pos_kernel<<<(RCOLS * HD + 255) / 256, 256, 0, stream>>>(pos, posBf);

  gemm_qkv_kernel<<<dim3(TOK / 64, DMODEL / 64, 3), 128, 0, stream>>>(
      qryBf, wqT, wkT, wvT, qh, kh, vhT);

  attn_kernel<<<dim3(LSEQ / 16, BATCH * NHEADS), 32, 0, stream>>>(
      qh, kh, vhT, posBf, bias, attnBf);

  gemm_out_kernel<<<dim3(TOK / 64, DMODEL / 64), 128, 0, stream>>>(
      attnBf, woT, out);
}
